// LinearAttention_12008728560180
// MI455X (gfx1250) — matrix operand order repaired and checked by reading, not compiled, not run on hardware
//
#include <hip/hip_runtime.h>
#include <hip/hip_bf16.h>
#include <math.h>

// MI455X / gfx1250, wave32. bf16 WMMA (v_wmma_f32_16x16x32_bf16) with fp32 accum.
// Workspace requirement: ~51 MB (3 x 16MB bf16 Q/K/V + 1MB W2).

typedef __bf16 bf16;
typedef __attribute__((ext_vector_type(4)))  float   v4f;
typedef __attribute__((ext_vector_type(4)))  __bf16  v4bf;
typedef __attribute__((ext_vector_type(8)))  __bf16  v8bf;
typedef __attribute__((ext_vector_type(16))) __bf16  v16bf;
typedef __attribute__((ext_vector_type(8)))  float   v8f;

// Build a 16-element bf16 fragment (8 VGPRs) from two 16-byte chunks.
// A-matrix frag: chunks at +0 and +16 elements.  B-matrix frag: also +0 and +16 elements.
__device__ __forceinline__ v16bf load_frag(const bf16* lo, const bf16* hi) {
    v8bf a = *(const v8bf*)lo;
    v8bf b = *(const v8bf*)hi;
    v16bf r;
#pragma unroll
    for (int i = 0; i < 8; ++i) { r[i] = a[i]; r[i + 8] = b[i]; }
    return r;
}

__device__ __forceinline__ v8f wmma_bf16(v16bf a, v16bf b, v8f c) {
    // D = A(16x32 bf16) x B(32x16 bf16) + C(16x16 f32)
    return __builtin_amdgcn_wmma_f32_16x16x32_bf16(
        /*neg_a=*/false, a, /*neg_b=*/false, b,
        /*c_mod=*/(short)0, c, /*reuse_a=*/false, /*reuse_b=*/false);
}

__device__ __forceinline__ v4bf cvt4(float a, float b, float c, float d) {
    v4bf r; r[0] = (bf16)a; r[1] = (bf16)b; r[2] = (bf16)c; r[3] = (bf16)d;
    return r;
}

// ---------------------------------------------------------------------------
// Kernel 1: QKV GEMM (M=384 over 12 o-blocks of 32, K=256, N=4096 per batch)
// + per-block epilogue: q: softmax over d (32 rows) * scale; k: exp(); v: copy.
// Outputs bf16 to workspace, layout [b][128][4096] per tensor.
// ---------------------------------------------------------------------------
__global__ __launch_bounds__(256) void k_qkv(
    const float* __restrict__ x, const float* __restrict__ wqkv,
    bf16* __restrict__ Qg, bf16* __restrict__ Kg, bf16* __restrict__ Vg)
{
    const int nb  = blockIdx.x;         // 0..31  (128-column tile)
    const int ob  = blockIdx.y;         // 0..11  (32-row o-block)
    const int b   = blockIdx.z;         // 0..15
    const int t   = threadIdx.x;
    const int lane = t & 31, wave = t >> 5;
    const int n0  = nb * 128;

    __shared__ __align__(16) bf16  Aw[32 * 264];   // w block [32][256] pad 8
    __shared__ __align__(16) bf16  Bt[128 * 40];   // x tile transposed [n][k] pad 8
    __shared__ float Cw[8 * 32 * 17];              // per-wave 32x16 C, pad 17
    __shared__ __align__(16) bf16  Ew[32 * 136];   // bf16 results tile [32][128] pad 8

    // Stage weight block as bf16 (vectorized: float4 load + b64 LDS store)
#pragma unroll
    for (int p = 0; p < 8; ++p) {
        const int f = t + p * 256;
        const int r = f >> 6, c4 = (f & 63) << 2;
        v4f wv = *(const v4f*)&wqkv[(size_t)(ob * 32 + r) * 256 + c4];
        *(v4bf*)&Aw[r * 264 + c4] = cvt4(wv[0], wv[1], wv[2], wv[3]);
    }

    v8f c0 = {}; v8f c1 = {};
    const int mrow = lane & 15;
    const int aoff = (lane < 16) ? 0 : 8;    // A frag K sub-offset per lane half
    const int boff = (lane < 16) ? 0 : 8;    // B frag K sub-offset per lane half

    // Staging micro-tile: this thread owns k rows kst..kst+3 at cols nst..nst+3
    const int kst = wave << 2;               // 0..28
    const int nst = lane << 2;               // 0..124
    const float* xb = x + (size_t)b * 256 * 4096 + n0 + nst;

    // Preload x micro-tile for ks = 0
    v4f xr0 = *(const v4f*)&xb[(size_t)(kst + 0) * 4096];
    v4f xr1 = *(const v4f*)&xb[(size_t)(kst + 1) * 4096];
    v4f xr2 = *(const v4f*)&xb[(size_t)(kst + 2) * 4096];
    v4f xr3 = *(const v4f*)&xb[(size_t)(kst + 3) * 4096];

    for (int ks = 0; ks < 8; ++ks) {
        const int kb = ks * 32;
        // Convert + transposed store: 4 bf16 along k are contiguous -> b64 stores
#pragma unroll
        for (int j = 0; j < 4; ++j) {
            *(v4bf*)&Bt[(nst + j) * 40 + kst] = cvt4(xr0[j], xr1[j], xr2[j], xr3[j]);
        }
        __syncthreads();
        if (ks < 7) {  // issue next tile's loads early; waited one iteration later
            const float* xn = xb + (size_t)(kb + 32) * 4096;
            xr0 = *(const v4f*)&xn[(size_t)(kst + 0) * 4096];
            xr1 = *(const v4f*)&xn[(size_t)(kst + 1) * 4096];
            xr2 = *(const v4f*)&xn[(size_t)(kst + 2) * 4096];
            xr3 = *(const v4f*)&xn[(size_t)(kst + 3) * 4096];
        }
        v16bf a0 = load_frag(&Aw[(size_t)(mrow)      * 264 + kb + aoff],
                             &Aw[(size_t)(mrow)      * 264 + kb + aoff + 16]);
        v16bf a1 = load_frag(&Aw[(size_t)(mrow + 16) * 264 + kb + aoff],
                             &Aw[(size_t)(mrow + 16) * 264 + kb + aoff + 16]);
        const bf16* bp = &Bt[(size_t)(wave * 16 + mrow) * 40 + boff];
        v16bf bb = load_frag(bp, bp + 16);
        c0 = wmma_bf16(a0, bb, c0);
        c1 = wmma_bf16(a1, bb, c1);
        __syncthreads();
    }

    // Spill C tiles to LDS: VGPR r, lanes0-15 -> M=r, lanes16-31 -> M=r+8
    float* Cp = &Cw[wave * 32 * 17];
    const int mbase = (lane < 16) ? 0 : 8;
#pragma unroll
    for (int r = 0; r < 8; ++r) {
        Cp[(mbase + r)      * 17 + mrow] = c0[r];
        Cp[(mbase + r + 16) * 17 + mrow] = c1[r];
    }
    __syncthreads();

    if (lane < 16) {   // per-column epilogue -> bf16 results into Ew
        float col[32];
#pragma unroll
        for (int j = 0; j < 32; ++j) col[j] = Cp[j * 17 + lane];
        const int ecol = wave * 16 + lane;
        if (ob < 4) {                       // q: softmax over d, * DIM_HEAD^-0.5
            float m = col[0];
#pragma unroll
            for (int j = 1; j < 32; ++j) m = fmaxf(m, col[j]);
            float s = 0.f;
#pragma unroll
            for (int j = 0; j < 32; ++j) { col[j] = __expf(col[j] - m); s += col[j]; }
            const float inv = 0.17677669529663689f / s;
#pragma unroll
            for (int j = 0; j < 32; ++j) Ew[j * 136 + ecol] = (bf16)(col[j] * inv);
        } else if (ob < 8) {                // k: unnormalized exp (norm folded into ctx)
#pragma unroll
            for (int j = 0; j < 32; ++j) Ew[j * 136 + ecol] = (bf16)__expf(col[j]);
        } else {                            // v: passthrough
#pragma unroll
            for (int j = 0; j < 32; ++j) Ew[j * 136 + ecol] = (bf16)col[j];
        }
    }
    __syncthreads();

    // Cooperative coalesced store: 32 rows x 128 cols, 32B (2x b128) per thread
    {
        bf16* dst = (ob < 4) ? Qg : (ob < 8) ? Kg : Vg;
        const int j = t >> 3, nn = (t & 7) << 4;
        const size_t row = (size_t)b * 128 + (size_t)(ob & 3) * 32 + j;
        v8bf e0 = *(const v8bf*)&Ew[j * 136 + nn];
        v8bf e1 = *(const v8bf*)&Ew[j * 136 + nn + 8];
        *(v8bf*)&dst[row * 4096 + n0 + nn]     = e0;
        *(v8bf*)&dst[row * 4096 + n0 + nn + 8] = e1;
    }
}

// ---------------------------------------------------------------------------
// Kernel 2: per (b,h): ctx[d][e] = (sum_n expk[d][n]*v[e][n]) / (sum_n expk[d][n])
// then fold output projection: W2[o][h*32+d] = sum_e w_out[o][h*32+e] * ctx[d][e]
// 64 workgroups, 8 waves split K=4096.
// ---------------------------------------------------------------------------
__global__ __launch_bounds__(256) void k_ctx(
    const bf16* __restrict__ Kg, const bf16* __restrict__ Vg,
    const float* __restrict__ wout, bf16* __restrict__ W2g)
{
    const int bh = blockIdx.x;                 // 0..63
    const int b = bh >> 2, h = bh & 3;
    const int t = threadIdx.x;
    const int lane = t & 31, wave = t >> 5;

    __shared__ float ctxp[8 * 32 * 32];        // per-wave partial ctx
    __shared__ float ksump[8 * 2 * 32];        // per-wave per-lane k row-sum partials
    __shared__ float ctxf[32 * 32];
    __shared__ float ksum[32];

    const bf16* Kb = Kg + ((size_t)bh * 32) * 4096;
    const bf16* Vb = Vg + ((size_t)bh * 32) * 4096;

    v8f acc00 = {}, acc01 = {}, acc10 = {}, acc11 = {};
    float s0 = 0.f, s1 = 0.f;
    const int mrow = lane & 15;
    const int aoff = (lane < 16) ? 0 : 8;
    const int boff = (lane < 16) ? 0 : 8;

#pragma unroll 2
    for (int s = 0; s < 16; ++s) {
        const int nbs = wave * 512 + s * 32;
        const bf16* a0p = Kb + (size_t)mrow        * 4096 + nbs + aoff;
        const bf16* a1p = Kb + (size_t)(mrow + 16) * 4096 + nbs + aoff;
        v16bf a0 = load_frag(a0p, a0p + 16);
        v16bf a1 = load_frag(a1p, a1p + 16);
        const bf16* b0p = Vb + (size_t)mrow        * 4096 + nbs + boff;
        const bf16* b1p = Vb + (size_t)(mrow + 16) * 4096 + nbs + boff;
        v16bf b0 = load_frag(b0p, b0p + 16);
        v16bf b1 = load_frag(b1p, b1p + 16);
        acc00 = wmma_bf16(a0, b0, acc00);
        acc01 = wmma_bf16(a0, b1, acc01);
        acc10 = wmma_bf16(a1, b0, acc10);
        acc11 = wmma_bf16(a1, b1, acc11);
#pragma unroll
        for (int i = 0; i < 16; ++i) { s0 += (float)a0[i]; s1 += (float)a1[i]; }
    }

    float* cp = &ctxp[wave * 1024];
    const int d0 = (lane < 16) ? 0 : 8;
#pragma unroll
    for (int r = 0; r < 8; ++r) {
        cp[(d0 + r)      * 32 + mrow     ] = acc00[r];
        cp[(d0 + r)      * 32 + 16 + mrow] = acc01[r];
        cp[(d0 + r + 16) * 32 + mrow     ] = acc10[r];
        cp[(d0 + r + 16) * 32 + 16 + mrow] = acc11[r];
    }
    ksump[wave * 64 + lane]      = s0;   // rows 0..15 (lane&15), both K halves
    ksump[wave * 64 + 32 + lane] = s1;   // rows 16..31
    __syncthreads();

    if (t < 32) {  // total k row sums
        const int tile = t >> 4, r = t & 15;
        float s = 0.f;
#pragma unroll
        for (int w = 0; w < 8; ++w)
            s += ksump[w * 64 + tile * 32 + r] + ksump[w * 64 + tile * 32 + r + 16];
        ksum[t] = fmaxf(s, 1e-30f);
    }
    __syncthreads();
    for (int idx = t; idx < 1024; idx += 256) {
        const int d = idx >> 5;
        float s = 0.f;
#pragma unroll
        for (int w = 0; w < 8; ++w) s += ctxp[w * 1024 + idx];
        ctxf[idx] = s / ksum[d];
    }
    __syncthreads();

    // Fold with w_out: thread t owns output channel o = t
    {
        const int o = t;
        float wrow[32];
#pragma unroll
        for (int e = 0; e < 32; ++e) wrow[e] = wout[(size_t)o * 128 + h * 32 + e];
        for (int d = 0; d < 32; ++d) {
            float a = 0.f;
#pragma unroll
            for (int e = 0; e < 32; ++e) a += wrow[e] * ctxf[d * 32 + e];
            W2g[((size_t)b * 256 + o) * 128 + h * 32 + d] = (bf16)a;
        }
    }
}

// ---------------------------------------------------------------------------
// Kernel 3: per batch: out = W2_b(256x128) @ Q_b(128x4096) + b_out, then
// L2-normalize over channels, * g * 16. Workgroup = full M=256 x N=32 tile.
// ---------------------------------------------------------------------------
__global__ __launch_bounds__(256) void k_out(
    const bf16* __restrict__ W2g, const bf16* __restrict__ Qg,
    const float* __restrict__ bout, const float* __restrict__ g,
    float* __restrict__ out)
{
    const int nb = blockIdx.x;                // 0..127
    const int b  = blockIdx.y;                // 0..15
    const int t  = threadIdx.x;
    const int lane = t & 31, wave = t >> 5;
    const int n0 = nb * 32;

    __shared__ __align__(16) char ubuf[36864];   // A2s[256][40] bf16  /  Cs[256][33] f32
    bf16*  A2s = (bf16*)ubuf;
    float* Cs  = (float*)ubuf;
    __shared__ __align__(16) bf16 Btile[32 * 40];
    __shared__ float normp[256];
    __shared__ float biasS[256];
    __shared__ float gS[256];

    biasS[t] = bout[t];
    gS[t]    = g[t];

    const bf16* W2b = W2g + (size_t)b * 256 * 128;
    const bf16* Qb  = Qg  + (size_t)b * 128 * 4096;

    v8f acc00 = {}, acc01 = {}, acc10 = {}, acc11 = {};
    const int Mbase = wave * 32;
    const int mrow = lane & 15;
    const int aoff = (lane < 16) ? 0 : 8;
    const int boff = (lane < 16) ? 0 : 8;

    for (int ks = 0; ks < 4; ++ks) {
        const int kb = ks * 32;
        __syncthreads();
        // Stage A slab [256][kb..kb+31]: thread t copies its row (4 x b128)
        {
            const bf16* src = W2b + (size_t)t * 128 + kb;
#pragma unroll
            for (int cch = 0; cch < 4; ++cch)
                *(v8bf*)&A2s[t * 40 + cch * 8] = *(const v8bf*)&src[cch * 8];
        }
        // Stage transposed Q tile [n][k]: thread t: row k=t>>3, 4 cols
        {
            const int k = t >> 3, n4 = (t & 7) << 2;
            v4bf qv = *(const v4bf*)&Qb[(size_t)(kb + k) * 4096 + n0 + n4];
#pragma unroll
            for (int j = 0; j < 4; ++j) Btile[(n4 + j) * 40 + k] = qv[j];
        }
        __syncthreads();
        v16bf a0 = load_frag(&A2s[(Mbase + mrow)      * 40 + aoff],
                             &A2s[(Mbase + mrow)      * 40 + aoff + 16]);
        v16bf a1 = load_frag(&A2s[(Mbase + 16 + mrow) * 40 + aoff],
                             &A2s[(Mbase + 16 + mrow) * 40 + aoff + 16]);
        v16bf b0 = load_frag(&Btile[(mrow)      * 40 + boff], &Btile[(mrow)      * 40 + boff + 16]);
        v16bf b1 = load_frag(&Btile[(16 + mrow) * 40 + boff], &Btile[(16 + mrow) * 40 + boff + 16]);
        acc00 = wmma_bf16(a0, b0, acc00);
        acc01 = wmma_bf16(a0, b1, acc01);
        acc10 = wmma_bf16(a1, b0, acc10);
        acc11 = wmma_bf16(a1, b1, acc11);
    }
    __syncthreads();

    // C tiles -> Cs[256][33] (aliases A2s; all GEMM reads complete)
    const int mm = (lane < 16) ? 0 : 8;
#pragma unroll
    for (int r = 0; r < 8; ++r) {
        Cs[(Mbase + mm + r)      * 33 + mrow     ] = acc00[r];
        Cs[(Mbase + mm + r)      * 33 + 16 + mrow] = acc01[r];
        Cs[(Mbase + mm + r + 16) * 33 + mrow     ] = acc10[r];
        Cs[(Mbase + mm + r + 16) * 33 + 16 + mrow] = acc11[r];
    }
    __syncthreads();

    // Bias + channel-L2-norm epilogue. Thread t owns column n=t%32, rows grp*32..+31
    const int n = t & 31, grp = t >> 5, obase = grp * 32;
    float part = 0.f;
#pragma unroll
    for (int o = 0; o < 32; ++o) {
        float v = Cs[(obase + o) * 33 + n] + biasS[obase + o];
        Cs[(obase + o) * 33 + n] = v;
        part += v * v;
    }
    normp[grp * 32 + n] = part;
    __syncthreads();
    float nrm = 0.f;
#pragma unroll
    for (int w = 0; w < 8; ++w) nrm += normp[w * 32 + n];
    const float inv = 16.0f / fmaxf(sqrtf(nrm), 1e-12f);   // sqrt(c)=16
#pragma unroll
    for (int o = 0; o < 32; ++o) {
        out[((size_t)b * 256 + obase + o) * 4096 + n0 + n] =
            Cs[(obase + o) * 33 + n] * inv * gS[obase + o];
    }
}

// ---------------------------------------------------------------------------
extern "C" void kernel_launch(void* const* d_in, const int* in_sizes, int n_in,
                              void* d_out, int out_size, void* d_ws, size_t ws_size,
                              hipStream_t stream) {
    (void)in_sizes; (void)n_in; (void)out_size; (void)ws_size;
    const float* x    = (const float*)d_in[0];   // (16,256,64,64)
    const float* wqkv = (const float*)d_in[1];   // (384,256)
    const float* wout = (const float*)d_in[2];   // (256,128)
    const float* bout = (const float*)d_in[3];   // (256,)
    const float* g    = (const float*)d_in[4];   // (1,256,1,1)
    float* out = (float*)d_out;

    const size_t QKV_ELEMS = (size_t)16 * 128 * 4096;
    bf16* Qg  = (bf16*)d_ws;
    bf16* Kg  = Qg + QKV_ELEMS;
    bf16* Vg  = Kg + QKV_ELEMS;
    bf16* W2g = Vg + QKV_ELEMS;                  // 16*256*128 bf16

    k_qkv<<<dim3(32, 12, 16), 256, 0, stream>>>(x, wqkv, Qg, Kg, Vg);
    k_ctx<<<dim3(64), 256, 0, stream>>>(Kg, Vg, wout, W2g);
    k_out<<<dim3(128, 16), 256, 0, stream>>>(W2g, Qg, bout, g, out);
}
